// Updating_H_Layer_32074815766811
// MI455X (gfx1250) — compile-verified, matrix operand order checked by reading, not run on hardware
//
#include <hip/hip_runtime.h>
#include <hip/hip_bf16.h>
#include <stdint.h>
#include <stddef.h>

#define ROWS 4096
#define COLS 8192
#define RANK 16
#define MAX_ITER 50
#define ALPHA_C 0.1f
#define LAMBDA_PT 0.01f
#define K2_C 1.0f
#define TOL_C 1e-8f

typedef __attribute__((ext_vector_type(16))) __bf16 v16bf;
typedef __attribute__((ext_vector_type(8)))  __bf16 v8bf;
typedef __attribute__((ext_vector_type(4)))  __bf16 v4bf;
typedef __attribute__((ext_vector_type(8)))  float  v8f;

__device__ __forceinline__ v16bf cat8(v8bf lo, v8bf hi) {
  return __builtin_shufflevector(lo, hi, 0,1,2,3,4,5,6,7,8,9,10,11,12,13,14,15);
}
__device__ __forceinline__ v8bf ldg8(const __bf16* p) { return *(const v8bf*)p; }

// ---------------------------------------------------------------- TTt -> bf16
__global__ void k_cvt_ttt(const float* __restrict__ T, __bf16* __restrict__ Tb) {
  size_t base = ((size_t)blockIdx.x * 256 + threadIdx.x) * 4;
  float4 v = *(const float4*)(T + base);
  v4bf o;
  o[0] = (__bf16)v.x; o[1] = (__bf16)v.y; o[2] = (__bf16)v.z; o[3] = (__bf16)v.w;
  *(v4bf*)(Tb + base) = o;
}

// -------------------- OmT[j][i] = bf16(Omega[i][j])  (64x64 LDS tile transpose)
// grid (COLS/64, ROWS/64), block 256
__global__ void k_omt(const int* __restrict__ Om, __bf16* __restrict__ OmT) {
  __shared__ float tile[64 * 65];                   // padded: conflict-free both phases
  int j0 = blockIdx.x * 64;
  int i0 = blockIdx.y * 64;
  int tj = threadIdx.x & 63;                        // fast index within a 64-run
  int tq = threadIdx.x >> 6;                        // 0..3
  #pragma unroll
  for (int it = 0; it < 16; ++it) {                 // read rows of Omega (coalesced b32)
    int il = it * 4 + tq;
    tile[tj * 65 + il] = (float)Om[(size_t)(i0 + il) * COLS + j0 + tj];
  }
  __syncthreads();
  #pragma unroll
  for (int it = 0; it < 16; ++it) {                 // write rows of OmT (coalesced b16)
    int jl = it * 4 + tq;
    OmT[(size_t)(j0 + jl) * ROWS + i0 + tj] = (__bf16)tile[jl * 65 + tj];
  }
}

// ------------------------------------------- WpT[16r+s][i] = bf16(W[i,r]W[i,s])
__global__ void k_wpt(const float* __restrict__ W, __bf16* __restrict__ WpT) {
  int i = blockIdx.x * 256 + threadIdx.x;           // 0..4095
  float w[16];
  #pragma unroll
  for (int r = 0; r < 16; ++r) w[r] = W[i * 16 + r];
  #pragma unroll
  for (int r = 0; r < 16; ++r)
    #pragma unroll
    for (int s = 0; s < 16; ++s)
      WpT[(size_t)(r * 16 + s) * ROWS + i] = (__bf16)(w[r] * w[s]);
}

// ------------------------------------------------ U_H = W^T (Omega .* Z)
__global__ void k_uh(const int* __restrict__ Om, const float* __restrict__ Z,
                     const float* __restrict__ W, float* __restrict__ U) {
  __shared__ float lw[128 * 16];
  int j = blockIdx.x * 256 + threadIdx.x;           // column
  float acc[16] = {};
  for (int c = 0; c < ROWS / 128; ++c) {
    __syncthreads();
    for (int e = threadIdx.x; e < 128 * 16; e += 256) lw[e] = W[c * 2048 + e];
    __syncthreads();
    #pragma unroll 4
    for (int ii = 0; ii < 128; ++ii) {
      int i = c * 128 + ii;
      int   om = Om[(size_t)i * COLS + j];
      float z  = Z [(size_t)i * COLS + j];
      float mz = om ? z : 0.0f;
      #pragma unroll
      for (int r = 0; r < 16; ++r) acc[r] = fmaf(mz, lw[ii * 16 + r], acc[r]);
    }
  }
  #pragma unroll
  for (int r = 0; r < 16; ++r) U[(size_t)r * COLS + j] = acc[r];
}

// ---------------- L_H = Omega^T @ Wp  via v_wmma_f32_16x16x32_bf16 (+ alpha*I)
// grid (512 j-tiles, 4), block 128 = 4 waves, wave w -> p-tile (r-index) pt.
// Pure load/wmma loop: no LDS, no barriers (OmT & WpT pre-laid-out as B^T rows).
__global__ void k_lh(const __bf16* __restrict__ OmT, const __bf16* __restrict__ WpT,
                     float* __restrict__ LH) {
  int j0   = blockIdx.x * 16;
  int wave = threadIdx.x >> 5;
  int lane = threadIdx.x & 31;
  int pt   = blockIdx.y * 4 + wave;                 // 0..15 == r
  int n    = lane & 15, half = lane >> 4;
  const __bf16* arow = OmT + (size_t)(j0 + n) * ROWS;        // A row m (=lane&15)
  const __bf16* brow = WpT + (size_t)(pt * 16 + n) * ROWS;   // B^T row n
  v8f c = {};
  for (int k = 0; k < ROWS; k += 32) {
    __builtin_prefetch(brow + k + 512, 0, 0);
    v16bf a = cat8(ldg8(arow + k + 8 * half), ldg8(arow + k + 16 + 8 * half));
    v16bf b = cat8(ldg8(brow + k + 8 * half), ldg8(brow + k + 16 + 8 * half));
    c = __builtin_amdgcn_wmma_f32_16x16x32_bf16(false, a, false, b, (short)0, c, false, false);
  }
  #pragma unroll
  for (int v = 0; v < 8; ++v) {
    int m = v + 8 * half;                           // C row = j offset
    int s = n;                                      // C col = s index
    float diag = (s == pt) ? ALPHA_C : 0.0f;        // + alpha * I_r
    LH[(size_t)(j0 + m) * 256 + pt * 16 + s] = c[v] + diag;
  }
}

// ------------------------- lambda_j = max eig of L_H[j] (power iteration)
__global__ void k_lam(const float* __restrict__ LH, float* __restrict__ lam) {
  __shared__ float lm[64 * 257];                    // padded: conflict-free
  int t = threadIdx.x;
  int j = blockIdx.x * 64 + t;
  const float* M = LH + (size_t)j * 256;
  for (int e = 0; e < 256; ++e) lm[t * 257 + e] = M[e];
  float v[16];
  #pragma unroll
  for (int r = 0; r < 16; ++r) v[r] = 0.25f;        // unit start vector
  float nrm = ALPHA_C;
  for (int it = 0; it < 48; ++it) {
    float w[16];
    #pragma unroll
    for (int r = 0; r < 16; ++r) {
      float s = 0.0f;
      #pragma unroll
      for (int q = 0; q < 16; ++q) s = fmaf(lm[t * 257 + r * 16 + q], v[q], s);
      w[r] = s;
    }
    float ss = 0.0f;
    #pragma unroll
    for (int r = 0; r < 16; ++r) ss = fmaf(w[r], w[r], ss);
    nrm = sqrtf(ss);
    float inv = 1.0f / nrm;
    #pragma unroll
    for (int r = 0; r < 16; ++r) v[r] = w[r] * inv;
  }
  lam[j] = nrm;                                     // PSD+alpha*I -> nrm >= alpha
}

// ------------------------------------------------ init H buffers + scalars
__global__ void k_init(const float* __restrict__ Hp, float* __restrict__ H,
                       __bf16* __restrict__ Hb, float* accum, float* term, int* active) {
  int idx = blockIdx.x * 256 + threadIdx.x;
  float h = Hp[idx];
  H[idx]  = h;
  Hb[idx] = (__bf16)h;
  if (idx == 0) { *accum = 0.0f; *term = 1.0f; *active = 1; }
}

// ---- one ISTA step: P = H@TTt (WMMA, K split over 8 waves) + fused epilogue
// grid 512 (16-column tiles), block 256 (8 waves)
template <bool FINAL>
__global__ void k_iter(const __bf16* __restrict__ Tb, const __bf16* __restrict__ Hb,
                       const float* __restrict__ H,  const float* __restrict__ LH,
                       const float* __restrict__ U,  const float* __restrict__ lam,
                       float* __restrict__ Hout, __bf16* __restrict__ Hbout,
                       const int* __restrict__ active, float* __restrict__ accum) {
  __shared__ float ldsC[8 * 256];                   // per-wave partial C tiles
  __shared__ float ldsH[256];                       // H column slab (fp32)
  int j0   = blockIdx.x * 16;
  int wave = threadIdx.x >> 5;
  int lane = threadIdx.x & 31;
  int n    = lane & 15, half = lane >> 4;
  const __bf16* arow = Hb + (size_t)n * COLS;                // A row m (=lane&15)
  const __bf16* brow = Tb + (size_t)(j0 + n) * COLS;         // TTt symmetric: row == B^T row
  v8f c = {};
  int kbeg = wave * (COLS / 8);
  int kend = kbeg + (COLS / 8);
  for (int k = kbeg; k < kend; k += 32) {
    __builtin_prefetch(brow + k + 512, 0, 0);                // global_prefetch_b8
    v16bf a = cat8(ldg8(arow + k + 8 * half), ldg8(arow + k + 16 + 8 * half));
    v16bf b = cat8(ldg8(brow + k + 8 * half), ldg8(brow + k + 16 + 8 * half));
    c = __builtin_amdgcn_wmma_f32_16x16x32_bf16(false, a, false, b, (short)0, c, false, false);
  }
  #pragma unroll
  for (int v = 0; v < 8; ++v) ldsC[wave * 256 + v * 32 + lane] = c[v];
  {
    int s = threadIdx.x >> 4, nn = threadIdx.x & 15;
    ldsH[s * 16 + nn] = H[(size_t)s * COLS + j0 + nn];
  }
  __syncthreads();

  int m = threadIdx.x >> 4, nn = threadIdx.x & 15;           // (rank r, col-in-tile)
  int j = j0 + nn;
  float p = 0.0f;                                            // (H@TTt)[m, j]
  #pragma unroll
  for (int w = 0; w < 8; ++w)
    p += ldsC[w * 256 + (m & 7) * 32 + ((m >> 3) << 4) + nn];
  float lsum = 0.0f;                                         // (L_H[j] H[:,j])[m]
  const float* Mrow = LH + (size_t)j * 256 + m * 16;
  #pragma unroll
  for (int s = 0; s < 16; ++s) lsum = fmaf(Mrow[s], ldsH[s * 16 + nn], lsum);
  float hv = ldsH[m * 16 + nn];
  float l  = lam[j];
  float u  = U[(size_t)m * COLS + j];
  // relu(-S/2) = relu((U + lam*H - L_H H - 0.01*H TTt) / lam)
  float hn = fmaxf((u + l * hv - lsum - LAMBDA_PT * p) / l, 0.0f);
  if (FINAL) {
    Hout[(size_t)m * COLS + j] = K2_C * hn;                  // unconditional last step
  } else {
    if (!*active) hn = hv;                                   // freeze (sticky)
    Hout [(size_t)m * COLS + j] = hn;
    Hbout[(size_t)m * COLS + j] = (__bf16)hn;
    float d = hn - hv;
    __syncthreads();
    ldsC[threadIdx.x] = d * d;
    __syncthreads();
    for (int s = 128; s > 0; s >>= 1) {
      if (threadIdx.x < s) ldsC[threadIdx.x] += ldsC[threadIdx.x + s];
      __syncthreads();
    }
    if (threadIdx.x == 0) atomicAdd(accum, ldsC[0]);
  }
}

// ---------------------------------------- term / active update between steps
__global__ void k_fin(float* accum, float* term, int* active) {
  if (*active) {
    float t = *accum / (float)(RANK * COLS);
    *term = t;
    *active = (t > TOL_C) ? 1 : 0;
  }
  *accum = 0.0f;
}

extern "C" void kernel_launch(void* const* d_in, const int* in_sizes, int n_in,
                              void* d_out, int out_size, void* d_ws, size_t ws_size,
                              hipStream_t stream) {
  const int*   Om  = (const int*)d_in[0];     // Omega [4096,8192] int32
  const float* W   = (const float*)d_in[1];   // [4096,16]
  const float* Hp  = (const float*)d_in[2];   // [16,8192]
  const float* Z   = (const float*)d_in[3];   // [4096,8192]
  const float* TTt = (const float*)d_in[4];   // [8192,8192] symmetric
  // d_in[5] = I_r (folded into k_lh diag)

  char* ws = (char*)d_ws;
  size_t off = 0;
  auto alloc = [&](size_t bytes) -> void* {
    void* p = ws + off; off += (bytes + 255) & ~(size_t)255; return p;
  };
  __bf16* Tb    = (__bf16*)alloc((size_t)COLS * COLS * 2);     // 134 MB (L2-resident)
  __bf16* OmT   = (__bf16*)alloc((size_t)COLS * ROWS * 2);     // 67 MB
  __bf16* WpT   = (__bf16*)alloc((size_t)256 * ROWS * 2);      // 2 MB
  float*  U     = (float*) alloc((size_t)RANK * COLS * 4);     // 512 KB
  float*  LH    = (float*) alloc((size_t)COLS * 256 * 4);      // 8 MB
  float*  lamv  = (float*) alloc((size_t)COLS * 4);
  float*  Ha    = (float*) alloc((size_t)RANK * COLS * 4);
  float*  Hc2   = (float*) alloc((size_t)RANK * COLS * 4);
  __bf16* Hab   = (__bf16*)alloc((size_t)RANK * COLS * 2);
  __bf16* Hbb   = (__bf16*)alloc((size_t)RANK * COLS * 2);
  float*  accum = (float*) alloc(4);
  float*  term  = (float*) alloc(4);
  int*    act   = (int*)   alloc(4);

  k_cvt_ttt<<<(size_t)COLS * COLS / 1024, 256, 0, stream>>>(TTt, Tb);
  k_omt    <<<dim3(COLS / 64, ROWS / 64), 256, 0, stream>>>(Om, OmT);
  k_wpt    <<<ROWS / 256, 256, 0, stream>>>(W, WpT);
  k_uh     <<<COLS / 256, 256, 0, stream>>>(Om, Z, W, U);
  k_lh     <<<dim3(COLS / 16, 4), 128, 0, stream>>>(OmT, WpT, LH);
  k_lam    <<<COLS / 64, 64, 0, stream>>>(LH, lamv);
  k_init   <<<RANK * COLS / 256, 256, 0, stream>>>(Hp, Ha, Hab, accum, term, act);

  float*  Hcur = Ha;  float*  Hnxt = Hc2;
  __bf16* Hcb  = Hab; __bf16* Hnb  = Hbb;
  for (int it = 0; it < MAX_ITER - 1; ++it) {
    k_iter<false><<<COLS / 16, 256, 0, stream>>>(Tb, Hcb, Hcur, LH, U, lamv,
                                                 Hnxt, Hnb, act, accum);
    k_fin<<<1, 1, 0, stream>>>(accum, term, act);
    { float* t = Hcur; Hcur = Hnxt; Hnxt = t; }
    { __bf16* t = Hcb; Hcb = Hnb; Hnb = t; }
  }
  k_iter<true><<<COLS / 16, 256, 0, stream>>>(Tb, Hcb, Hcur, LH, U, lamv,
                                              (float*)d_out, Hnb, act, accum);
}
